// DGCNCCCNet_4002909519995
// MI455X (gfx1250) — compile-verified, matrix operand order checked by reading, not run on hardware
//
#include <hip/hip_runtime.h>

typedef __attribute__((ext_vector_type(16))) _Float16 v16h;
typedef __attribute__((ext_vector_type(8)))  float    v8f;

#define EPS_MM 1e-5f
#define EPS_L2 1e-6f

// K index held by fragment element e for this lane's K-group (wave32 WMMA
// 16-bit A 16x32 layout: lanes 0-15 hold K {0..7,16..23}, lanes 16-31 hold
// K {8..15,24..31}; element e = vgpr*2+half).
__device__ __forceinline__ int frag_k(int e, int kgrp) {
  return ((e >= 8) ? 16 : 0) + (kgrp << 3) + (e & 7);
}

// ---------------------------------------------------------------------------
// 3x3 conv as implicit GEMM on v_wmma_f32_16x16x32_f16.
// One wave -> 16 output pixels x 32 output channels (2 N-tiles share the A
// fragment). K is tap-major (k = tap*Cin + ci): 3x3 geometry hoisted to the
// 9-iteration outer loop. Bounds handled arithmetically (clamped-safe bases +
// 0/1 masks). Per-element addresses are strength-reduced: 16 hoisted per-lane
// offsets + a 32-bit incremental base (no 64-bit multiplies in the hot loop).
// ---------------------------------------------------------------------------
__global__ void __launch_bounds__(32)
conv2d_wmma(const float* __restrict__ in, const float* __restrict__ w,
            const float* __restrict__ bias, float* __restrict__ out,
            int Bn, int Cin, int H, int W, int Cout, int Ho, int Wo,
            int stride, int pad, int relu)
{
  const int lane = threadIdx.x & 31;
  const int row  = lane & 15;
  const int kgrp = lane >> 4;
  const int HW   = H * W;
  const int OHW  = Ho * Wo;
  const int M    = Bn * OHW;

  const int  m   = blockIdx.x * 16 + row;     // A-matrix row (output pixel)
  const bool mOK = (m < M);
  const int  mm  = mOK ? m : 0;
  const int  b   = mm / OHW;
  const int  r0  = mm - b * OHW;
  const int  oy  = r0 / Wo;
  const int  ox  = r0 - oy * Wo;

  const int   co0 = blockIdx.y * 32 + row;    // B columns (out channels)
  const int   co1 = co0 + 16;
  const float mk0 = (co0 < Cout) ? 1.f : 0.f;
  const float mk1 = (co1 < Cout) ? 1.f : 0.f;
  const float* inB = in + (size_t)b * Cin * HW;
  const float* w0  = w + (size_t)((co0 < Cout) ? co0 : 0) * Cin * 9;
  const float* w1  = w + (size_t)((co1 < Cout) ? co1 : 0) * Cin * 9;

  // Hoisted per-lane fragment offsets (32-bit; max Cin*HW ~ 7.5M < 2^31).
  int offA[16], offW[16];
#pragma unroll
  for (int e = 0; e < 16; ++e) {
    const int fk = frag_k(e, kgrp);
    offA[e] = fk * HW;
    offW[e] = fk * 9;
  }

  v8f acc0 = {0.f, 0.f, 0.f, 0.f, 0.f, 0.f, 0.f, 0.f};
  v8f acc1 = {0.f, 0.f, 0.f, 0.f, 0.f, 0.f, 0.f, 0.f};

  const int CinMain = Cin & ~31;              // full 32-wide K chunks
  const int stepA   = 32 * HW;

  for (int t = 0; t < 9; ++t) {               // 3x3 taps (outer, cheap)
    const int   ky  = t / 3;
    const int   kx  = t - ky * 3;
    const int   iy  = oy * stride + ky - pad;
    const int   ix  = ox * stride + kx - pad;
    const bool  tOK = mOK && (unsigned)iy < (unsigned)H && (unsigned)ix < (unsigned)W;
    const float mkT = tOK ? 1.f : 0.f;
    const float* inT = inB + (tOK ? (iy * W + ix) : 0);   // clamped-safe base
    const float* wt0 = w0 + t;
    const float* wt1 = w1 + t;

    int c0 = 0, baseA = 0, baseW = 0;
    for (; c0 < CinMain; c0 += 32, baseA += stepA, baseW += 288) {
      if (c0 + 32 < Cin)                       // uniform scalar condition
        __builtin_prefetch(&inT[baseA + stepA + offA[0]], 0, 1);
      v16h af, bf0, bf1;
#pragma unroll
      for (int e = 0; e < 16; ++e) {
        af[e]  = (_Float16)(inT[baseA + offA[e]] * mkT);
        bf0[e] = (_Float16)(wt0[baseW + offW[e]] * mk0);
        bf1[e] = (_Float16)(wt1[baseW + offW[e]] * mk1);
      }
      acc0 = __builtin_amdgcn_wmma_f32_16x16x32_f16(false, af, false, bf0,
                                                    (short)0, acc0, false, false);
      acc1 = __builtin_amdgcn_wmma_f32_16x16x32_f16(false, af, false, bf1,
                                                    (short)0, acc1, false, false);
    }
    if (c0 < Cin) {                            // masked tail chunk (cold)
      v16h af, bf0, bf1;
#pragma unroll
      for (int e = 0; e < 16; ++e) {
        const int   ci  = c0 + frag_k(e, kgrp);
        const bool  ok  = (ci < Cin);
        const int   ia  = ok ? (baseA + offA[e]) : 0;
        const int   iw  = ok ? (baseW + offW[e]) : 0;
        const float mkC = ok ? 1.f : 0.f;
        af[e]  = (_Float16)(inT[ia] * (mkT * mkC));
        bf0[e] = (_Float16)(wt0[iw] * (mk0 * mkC));
        bf1[e] = (_Float16)(wt1[iw] * (mk1 * mkC));
      }
      acc0 = __builtin_amdgcn_wmma_f32_16x16x32_f16(false, af, false, bf0,
                                                    (short)0, acc0, false, false);
      acc1 = __builtin_amdgcn_wmma_f32_16x16x32_f16(false, af, false, bf1,
                                                    (short)0, acc1, false, false);
    }
  }

  // C/D layout: lane holds column n = lane&15; VGPR r holds M = 8*kgrp + r.
  const int n = lane & 15;
  {
    const int oc = blockIdx.y * 32 + n;
    if (oc < Cout) {
      const float bsv = bias[oc];
#pragma unroll
      for (int r = 0; r < 8; ++r) {
        const int mo = blockIdx.x * 16 + kgrp * 8 + r;
        if (mo < M) {
          const int bb = mo / OHW;
          const int pp = mo - bb * OHW;
          float v = acc0[r] + bsv;
          if (relu) v = fmaxf(v, 0.f);
          out[((size_t)bb * Cout + oc) * OHW + pp] = v;
        }
      }
    }
  }
  {
    const int oc = blockIdx.y * 32 + 16 + n;
    if (oc < Cout) {
      const float bsv = bias[oc];
#pragma unroll
      for (int r = 0; r < 8; ++r) {
        const int mo = blockIdx.x * 16 + kgrp * 8 + r;
        if (mo < M) {
          const int bb = mo / OHW;
          const int pp = mo - bb * OHW;
          float v = acc1[r] + bsv;
          if (relu) v = fmaxf(v, 0.f);
          out[((size_t)bb * Cout + oc) * OHW + pp] = v;
        }
      }
    }
  }
}

// ---------------------------------------------------------------------------
// Correlation einsum 'bcij,bckl->b(ij)(kl)' as WMMA GEMM over 512 channels.
// One wave -> 16 rows x 32 cols (2 N-tiles share the A fragment). Clamped
// addresses + masks; strength-reduced addressing in the hot loop.
// ---------------------------------------------------------------------------
__global__ void __launch_bounds__(32)
corr_wmma(const float* __restrict__ fA, const float* __restrict__ fB,
          float* __restrict__ out, int C, int P)
{
  const int lane = threadIdx.x & 31;
  const int row  = lane & 15;
  const int kgrp = lane >> 4;
  const int b    = blockIdx.z;
  const int m    = blockIdx.x * 16 + row;
  const int n0   = blockIdx.y * 32 + row;
  const int n1   = n0 + 16;
  const int   mS  = (m  < P) ? m  : 0;  const float mkM  = (m  < P) ? 1.f : 0.f;
  const int   n0S = (n0 < P) ? n0 : 0;  const float mkN0 = (n0 < P) ? 1.f : 0.f;
  const int   n1S = (n1 < P) ? n1 : 0;  const float mkN1 = (n1 < P) ? 1.f : 0.f;
  const float* Am = fA + (size_t)b * C * P + mS;
  const float* B0 = fB + (size_t)b * C * P + n0S;
  const float* B1 = fB + (size_t)b * C * P + n1S;

  int offE[16];
#pragma unroll
  for (int e = 0; e < 16; ++e) offE[e] = frag_k(e, kgrp) * P;

  v8f acc0 = {0.f, 0.f, 0.f, 0.f, 0.f, 0.f, 0.f, 0.f};
  v8f acc1 = {0.f, 0.f, 0.f, 0.f, 0.f, 0.f, 0.f, 0.f};

  const int CMain = C & ~31;
  const int stepK = 32 * P;
  int k0 = 0, base = 0;
  for (; k0 < CMain; k0 += 32, base += stepK) {
    v16h af, bf0, bf1;
#pragma unroll
    for (int e = 0; e < 16; ++e) {
      af[e]  = (_Float16)(Am[base + offE[e]] * mkM);
      bf0[e] = (_Float16)(B0[base + offE[e]] * mkN0);
      bf1[e] = (_Float16)(B1[base + offE[e]] * mkN1);
    }
    acc0 = __builtin_amdgcn_wmma_f32_16x16x32_f16(false, af, false, bf0,
                                                  (short)0, acc0, false, false);
    acc1 = __builtin_amdgcn_wmma_f32_16x16x32_f16(false, af, false, bf1,
                                                  (short)0, acc1, false, false);
  }
  if (k0 < C) {
    v16h af, bf0, bf1;
#pragma unroll
    for (int e = 0; e < 16; ++e) {
      const int   k   = k0 + frag_k(e, kgrp);
      const bool  ok  = (k < C);
      const int   ik  = ok ? (base + offE[e]) : 0;
      const float mkK = ok ? 1.f : 0.f;
      af[e]  = (_Float16)(Am[ik] * (mkM * mkK));
      bf0[e] = (_Float16)(B0[ik] * (mkN0 * mkK));
      bf1[e] = (_Float16)(B1[ik] * (mkN1 * mkK));
    }
    acc0 = __builtin_amdgcn_wmma_f32_16x16x32_f16(false, af, false, bf0,
                                                  (short)0, acc0, false, false);
    acc1 = __builtin_amdgcn_wmma_f32_16x16x32_f16(false, af, false, bf1,
                                                  (short)0, acc1, false, false);
  }

  const int n = lane & 15;
#pragma unroll
  for (int r = 0; r < 8; ++r) {
    const int mo = blockIdx.x * 16 + kgrp * 8 + r;
    if (mo < P) {
      const int q0 = blockIdx.y * 32 + n;
      const int q1 = q0 + 16;
      if (q0 < P) out[((size_t)b * P + mo) * P + q0] = acc0[r];
      if (q1 < P) out[((size_t)b * P + mo) * P + q1] = acc1[r];
    }
  }
}

// ---------------------------------------------------------------------------
// L2 normalization over channel dim (optionally relu first); output channel
// stride may differ (used to write straight into a concat buffer).
// ---------------------------------------------------------------------------
__global__ void l2norm_kernel(const float* __restrict__ in, float* __restrict__ out,
                              int Bn, int C, int P, int CstrideOut, int relu)
{
  int idx = blockIdx.x * blockDim.x + threadIdx.x;
  if (idx >= Bn * P) return;
  int b = idx / P, p = idx % P;
  const float* ip = in + (size_t)b * C * P + p;
  float s = 0.f;
  for (int c = 0; c < C; ++c) {
    float v = ip[(size_t)c * P];
    if (relu) v = fmaxf(v, 0.f);
    s += v * v;
  }
  float inv = rsqrtf(s + EPS_L2);
  float* op = out + (size_t)b * CstrideOut * P + p;
  for (int c = 0; c < C; ++c) {
    float v = ip[(size_t)c * P];
    if (relu) v = fmaxf(v, 0.f);
    op[(size_t)c * P] = v * inv;
  }
}

// ------------------------- mutual matching ---------------------------------
__global__ void mm_max_kernel(const float* __restrict__ c, float* __restrict__ maxA,
                              float* __restrict__ maxB, int Bn, int P)
{
  int idx = blockIdx.x * blockDim.x + threadIdx.x;
  if (idx >= Bn * P) return;
  int b = idx / P, t = idx % P;
  const float* cb = c + (size_t)b * P * P;
  float mA = -3.4e38f, mB = -3.4e38f;
  for (int q = 0; q < P; ++q) {
    mA = fmaxf(mA, cb[(size_t)t * P + q]);   // max over kl for ij=t
    mB = fmaxf(mB, cb[(size_t)q * P + t]);   // max over ij for kl=t
  }
  maxA[idx] = mA;
  maxB[idx] = mB;
}

__global__ void mm_scale_kernel(const float* __restrict__ c, const float* __restrict__ maxA,
                                const float* __restrict__ maxB, float* __restrict__ out,
                                int Bn, int P)
{
  int idx = blockIdx.x * blockDim.x + threadIdx.x;
  if (idx >= Bn * P * P) return;
  int b = idx / (P * P);
  int r = (idx / P) % P;
  int q = idx % P;
  float v = c[idx];
  out[idx] = v * (v / (maxA[b * P + r] + EPS_MM)) * (v / (maxB[b * P + q] + EPS_MM));
}

// ---------------------- 4D conv (3^4 taps, SAME) ---------------------------
__global__ void conv4d_kernel(const float* __restrict__ in, const float* __restrict__ w,
                              const float* __restrict__ bias, float* __restrict__ out,
                              int Bn, int Cin, int Cout, int F, int relu)
{
  const int F2 = F * F, F3 = F2 * F, F4 = F2 * F2;
  int idx = blockIdx.x * blockDim.x + threadIdx.x;
  int total = Bn * Cout * F4;
  if (idx >= total) return;
  int b  = idx / (Cout * F4);
  int r  = idx % (Cout * F4);
  int co = r / F4;
  int p  = r % F4;
  int i  = p / F3;  int r2 = p % F3;
  int j  = r2 / F2; int r3 = r2 % F2;
  int k  = r3 / F;  int l  = r3 % F;
  float acc = bias[co];
  for (int ci = 0; ci < Cin; ++ci) {
    const float* ib = in + ((size_t)b * Cin + ci) * F4;
    const float* wb = w + ((size_t)co * Cin + ci) * 81;
    for (int ti = 0; ti < 3; ++ti) {
      int ii = i + ti - 1; if ((unsigned)ii >= (unsigned)F) continue;
      for (int tj = 0; tj < 3; ++tj) {
        int jj = j + tj - 1; if ((unsigned)jj >= (unsigned)F) continue;
        for (int tk = 0; tk < 3; ++tk) {
          int kk = k + tk - 1; if ((unsigned)kk >= (unsigned)F) continue;
          for (int tl = 0; tl < 3; ++tl) {
            int ll = l + tl - 1; if ((unsigned)ll >= (unsigned)F) continue;
            acc += ib[(((size_t)ii * F + jj) * F + kk) * F + ll] *
                   wb[((ti * 3 + tj) * 3 + tk) * 3 + tl];
          }
        }
      }
    }
  }
  if (relu) acc = fmaxf(acc, 0.f);
  out[idx] = acc;
}

// ---------------------- permutes / elementwise -----------------------------
__global__ void transpose_pp_kernel(const float* __restrict__ in, float* __restrict__ out,
                                    int Bn, int P)
{
  int idx = blockIdx.x * blockDim.x + threadIdx.x;
  if (idx >= Bn * P * P) return;
  int b = idx / (P * P);
  int r = (idx / P) % P;
  int q = idx % P;
  out[idx] = in[((size_t)b * P + q) * P + r];
}

__global__ void add_transposed_kernel(const float* __restrict__ a, const float* __restrict__ bt,
                                      float* __restrict__ out, int Bn, int P)
{
  int idx = blockIdx.x * blockDim.x + threadIdx.x;
  if (idx >= Bn * P * P) return;
  int b = idx / (P * P);
  int r = (idx / P) % P;
  int q = idx % P;
  out[idx] = a[idx] + bt[((size_t)b * P + q) * P + r];
}

__global__ void add_kernel(const float* __restrict__ a, const float* __restrict__ b,
                           float* __restrict__ out, int n)
{
  int idx = blockIdx.x * blockDim.x + threadIdx.x;
  if (idx < n) out[idx] = a[idx] + b[idx];
}

__global__ void zero_ch_kernel(float* __restrict__ buf, int Bn, int Ctot, int Cstart, int P)
{
  int nz = Ctot - Cstart;
  int idx = blockIdx.x * blockDim.x + threadIdx.x;
  if (idx >= Bn * nz * P) return;
  int b = idx / (nz * P);
  int r = idx % (nz * P);
  int c = r / P, p = r % P;
  buf[((size_t)b * Ctot + Cstart + c) * P + p] = 0.f;
}

__global__ void build_y_kernel(const float* __restrict__ p1w, const float* __restrict__ p2l,
                               const float* __restrict__ fu, float* __restrict__ y,
                               int Bn, int C, int P)
{
  int Ct = C + 2;
  int idx = blockIdx.x * blockDim.x + threadIdx.x;
  if (idx >= Bn * Ct * P) return;
  int b = idx / (Ct * P);
  int r = idx % (Ct * P);
  int c = r / P, p = r % P;
  float v;
  if (c < C)
    v = p1w[((size_t)b * C + c) * P + p] - p2l[((size_t)b * C + c) * P + p];
  else
    v = fu[((size_t)b * 2 + (c - C)) * P + p];
  y[idx] = v;
}

// ---------------------- bilinear upsample x2 (half-pixel) ------------------
__global__ void upsample2_kernel(const float* __restrict__ in, float* __restrict__ out,
                                 int Bn, int C, int h, int w)
{
  const int H = 2 * h, W = 2 * w;
  int idx = blockIdx.x * blockDim.x + threadIdx.x;
  int total = Bn * C * H * W;
  if (idx >= total) return;
  int p  = idx % (H * W);
  int bc = idx / (H * W);
  int oy = p / W, ox = p % W;
  float sy = (oy + 0.5f) * 0.5f - 0.5f;
  float sx = (ox + 0.5f) * 0.5f - 0.5f;
  float y0f = floorf(sy), x0f = floorf(sx);
  float fy = sy - y0f, fx = sx - x0f;
  int y0 = (int)y0f, x0 = (int)x0f;
  int y0c = y0 < 0 ? 0 : (y0 > h - 1 ? h - 1 : y0);
  int y1  = y0 + 1;
  int y1c = y1 < 0 ? 0 : (y1 > h - 1 ? h - 1 : y1);
  int x0c = x0 < 0 ? 0 : (x0 > w - 1 ? w - 1 : x0);
  int x1  = x0 + 1;
  int x1c = x1 < 0 ? 0 : (x1 > w - 1 ? w - 1 : x1);
  const float* ip = in + (size_t)bc * h * w;
  float v = ip[y0c * w + x0c] * (1.f - fy) * (1.f - fx)
          + ip[y0c * w + x1c] * (1.f - fy) * fx
          + ip[y1c * w + x0c] * fy * (1.f - fx)
          + ip[y1c * w + x1c] * fy * fx;
  out[idx] = v;
}

// ---------------------- grid sample (zeros padding) ------------------------
__device__ __forceinline__ float gs_tap(const float* ip, float yf, float xf, int H, int W)
{
  bool valid = (xf >= 0.f) && (xf <= (float)(W - 1)) && (yf >= 0.f) && (yf <= (float)(H - 1));
  int xi = (int)fminf(fmaxf(xf, 0.f), (float)(W - 1));
  int yi = (int)fminf(fmaxf(yf, 0.f), (float)(H - 1));
  float mask = valid ? 1.f : 0.f;
  return ip[yi * W + xi] * mask;   // clamped-safe load, arithmetic predication
}

__global__ void grid_sample_kernel(const float* __restrict__ img, const float* __restrict__ flow,
                                   float* __restrict__ out, int Bn, int C, int H, int W)
{
  const int Pp = H * W;
  int idx = blockIdx.x * blockDim.x + threadIdx.x;
  if (idx >= Bn * Pp) return;
  int b = idx / Pp, p = idx % Pp;
  float gx = (flow[((size_t)b * 2 + 0) * Pp + p] + 1.f) * 0.5f * (float)(W - 1);
  float gy = (flow[((size_t)b * 2 + 1) * Pp + p] + 1.f) * 0.5f * (float)(H - 1);
  float x0 = floorf(gx), y0 = floorf(gy);
  float x1 = x0 + 1.f,  y1 = y0 + 1.f;
  float wx1 = gx - x0, wx0 = 1.f - wx1;
  float wy1 = gy - y0, wy0 = 1.f - wy1;
  for (int c = 0; c < C; ++c) {
    const float* ip = img + ((size_t)b * C + c) * Pp;
    float v = gs_tap(ip, y0, x0, H, W) * wy0 * wx0
            + gs_tap(ip, y0, x1, H, W) * wy0 * wx1
            + gs_tap(ip, y1, x0, H, W) * wy1 * wx0
            + gs_tap(ip, y1, x1, H, W) * wy1 * wx1;
    out[((size_t)b * C + c) * Pp + p] = v;
  }
}

// ===========================================================================
extern "C" void kernel_launch(void* const* d_in, const int* in_sizes, int n_in,
                              void* d_out, int out_size, void* d_ws, size_t ws_size,
                              hipStream_t stream)
{
  (void)in_sizes; (void)n_in; (void)out_size; (void)ws_size;
  const float* x1 = (const float*)d_in[0];
  const float* x2 = (const float*)d_in[1];
  const float* pw[5];  const float* pb[5];
  for (int i = 0; i < 5; ++i) { pw[i] = (const float*)d_in[2 + 2 * i];  pb[i] = (const float*)d_in[3 + 2 * i]; }
  const float* nw[3];  const float* nb[3];
  for (int i = 0; i < 3; ++i) { nw[i] = (const float*)d_in[12 + 2 * i]; nb[i] = (const float*)d_in[13 + 2 * i]; }
  const float* r4w[4]; const float* r4b[4];
  for (int i = 0; i < 4; ++i) { r4w[i] = (const float*)d_in[18 + 2 * i]; r4b[i] = (const float*)d_in[19 + 2 * i]; }
  const float* ccw[5]; const float* ccb[5];
  for (int i = 0; i < 5; ++i) { ccw[i] = (const float*)d_in[26 + 2 * i]; ccb[i] = (const float*)d_in[27 + 2 * i]; }

  const int Bn = 2, F = 15, P = F * F;        // P = 225
  const int PH[5] = {240, 120, 60, 30, 15};
  const int PC[5] = {128, 128, 128, 128, 512};

  float* ws = (float*)d_ws;
  size_t off = 0;
  auto alloc = [&](size_t n) { float* p = ws + off; off += n; return p; };

  float* p1[5]; float* p2[5];
  for (int i = 0; i < 5; ++i) p1[i] = alloc((size_t)Bn * PC[i] * PH[i] * PH[i]);
  for (int i = 0; i < 5; ++i) p2[i] = alloc((size_t)Bn * PC[i] * PH[i] * PH[i]);
  float* FA    = alloc((size_t)Bn * 512 * P);
  float* FB    = alloc((size_t)Bn * 512 * P);
  float* CORR  = alloc((size_t)Bn * P * P);
  float* CMM   = alloc((size_t)Bn * P * P);
  float* CPERM = alloc((size_t)Bn * P * P);
  float* NT1   = alloc((size_t)Bn * 10 * P * P);
  float* NT2   = alloc((size_t)Bn * 10 * P * P);
  float* NA    = alloc((size_t)Bn * P * P);
  float* NB    = alloc((size_t)Bn * P * P);
  float* NRES  = alloc((size_t)Bn * P * P);
  float* CORRF = alloc((size_t)Bn * P * P);   // c_B_Avec view: (B,225,15,15)
  float* CABV  = alloc((size_t)Bn * P * P);   // c_A_Bvec
  float* MAXA  = alloc((size_t)Bn * P);
  float* MAXB  = alloc((size_t)Bn * P);
  float* X0    = alloc((size_t)Bn * 227 * P);
  float* FLOW  = alloc((size_t)Bn * 2 * 240 * 240);
  float* FU    = alloc((size_t)Bn * 2 * 240 * 240);
  float* DELTA = alloc((size_t)Bn * 2 * 240 * 240);
  float* BA    = alloc((size_t)15100000);     // ping (y / odd conv stages)
  float* BB    = alloc((size_t)15100000);     // pong (p1w / even conv stages)

  auto G1 = [](int n) { return dim3((unsigned)((n + 255) / 256)); };

  auto conv2d = [&](const float* in, const float* w, const float* bs, float* out,
                    int Cin, int H, int W, int Cout, int Ho, int Wo,
                    int stride, int pad, int relu) {
    dim3 g((unsigned)((Bn * Ho * Wo + 15) / 16), (unsigned)((Cout + 31) / 32), 1);
    conv2d_wmma<<<g, dim3(32), 0, stream>>>(in, w, bs, out, Bn, Cin, H, W, Cout,
                                            Ho, Wo, stride, pad, relu);
  };

  // ---- feature pyramids (SAME: pad 1 @ stride 1; pad_before 0 @ stride 2) --
  conv2d(x1, pw[0], pb[0], p1[0], 3, 240, 240, 128, 240, 240, 1, 1, 1);
  for (int i = 1; i < 5; ++i)
    conv2d(p1[i - 1], pw[i], pb[i], p1[i], 128, PH[i - 1], PH[i - 1], PC[i], PH[i], PH[i], 2, 0, 1);
  conv2d(x2, pw[0], pb[0], p2[0], 3, 240, 240, 128, 240, 240, 1, 1, 1);
  for (int i = 1; i < 5; ++i)
    conv2d(p2[i - 1], pw[i], pb[i], p2[i], 128, PH[i - 1], PH[i - 1], PC[i], PH[i], PH[i], 2, 0, 1);

  // ---- correlation + mutual matching + neighborhood consensus --------------
  l2norm_kernel<<<G1(Bn * P), 256, 0, stream>>>(p1[4], FA, Bn, 512, P, 512, 0);
  l2norm_kernel<<<G1(Bn * P), 256, 0, stream>>>(p2[4], FB, Bn, 512, P, 512, 0);
  corr_wmma<<<dim3((P + 15) / 16, (P + 31) / 32, Bn), dim3(32), 0, stream>>>(FA, FB, CORR, 512, P);

  auto mutual = [&](const float* src, float* dst) {
    mm_max_kernel<<<G1(Bn * P), 256, 0, stream>>>(src, MAXA, MAXB, Bn, P);
    mm_scale_kernel<<<G1(Bn * P * P), 256, 0, stream>>>(src, MAXA, MAXB, dst, Bn, P);
  };
  mutual(CORR, CMM);

  auto nc_apply = [&](const float* cin, float* outb) {
    conv4d_kernel<<<G1(Bn * 10 * P * P), 256, 0, stream>>>(cin, nw[0], nb[0], NT1, Bn, 1, 10, F, 1);
    conv4d_kernel<<<G1(Bn * 10 * P * P), 256, 0, stream>>>(NT1, nw[1], nb[1], NT2, Bn, 10, 10, F, 1);
    conv4d_kernel<<<G1(Bn * 1 * P * P), 256, 0, stream>>>(NT2, nw[2], nb[2], outb, Bn, 10, 1, F, 1);
  };
  transpose_pp_kernel<<<G1(Bn * P * P), 256, 0, stream>>>(CMM, CPERM, Bn, P);
  nc_apply(CMM, NA);
  nc_apply(CPERM, NB);
  add_transposed_kernel<<<G1(Bn * P * P), 256, 0, stream>>>(NA, NB, NRES, Bn, P);
  mutual(NRES, CORRF);
  transpose_pp_kernel<<<G1(Bn * P * P), 256, 0, stream>>>(CORRF, CABV, Bn, P);

  // ---- coarse-to-fine flow estimation --------------------------------------
  auto flow_est = [&](const float* corr, float* const* q1, float* const* q2, float* gout) {
    l2norm_kernel<<<G1(Bn * P), 256, 0, stream>>>(corr, X0, Bn, P, P, 227, 1);
    zero_ch_kernel<<<G1(Bn * 2 * P), 256, 0, stream>>>(X0, Bn, 227, 225, P);
    conv2d(X0, r4w[0], r4b[0], BA, 227, F, F, 128, F, F, 1, 1, 1);
    conv2d(BA, r4w[1], r4b[1], BB, 128, F, F, 64, F, F, 1, 1, 1);
    conv2d(BB, r4w[2], r4b[2], BA, 64, F, F, 32, F, F, 1, 1, 1);
    conv2d(BA, r4w[3], r4b[3], FLOW, 32, F, F, 2, F, F, 1, 1, 0);
    const float* flow = FLOW;
    int h = F;
    for (int lvl = 3; lvl >= 0; --lvl) {
      const int H2 = 2 * h;
      upsample2_kernel<<<G1(Bn * 2 * H2 * H2), 256, 0, stream>>>(flow, FU, Bn, 2, h, h);
      grid_sample_kernel<<<G1(Bn * H2 * H2), 256, 0, stream>>>(q1[lvl], FU, BB, Bn, 128, H2, H2);
      build_y_kernel<<<G1(Bn * 130 * H2 * H2), 256, 0, stream>>>(BB, q2[lvl], FU, BA, Bn, 128, H2 * H2);
      conv2d(BA, ccw[0], ccb[0], BB, 130, H2, H2, 128, H2, H2, 1, 1, 1);
      conv2d(BB, ccw[1], ccb[1], BA, 128, H2, H2, 96, H2, H2, 1, 1, 1);
      conv2d(BA, ccw[2], ccb[2], BB, 96, H2, H2, 64, H2, H2, 1, 1, 1);
      conv2d(BB, ccw[3], ccb[3], BA, 64, H2, H2, 32, H2, H2, 1, 1, 1);
      conv2d(BA, ccw[4], ccb[4], DELTA, 32, H2, H2, 2, H2, H2, 1, 1, 0);
      float* dst = (lvl == 0) ? gout : FLOW;
      add_kernel<<<G1(Bn * 2 * H2 * H2), 256, 0, stream>>>(FU, DELTA, dst, Bn * 2 * H2 * H2);
      flow = dst;
      h = H2;
    }
  };

  float* outf = (float*)d_out;
  flow_est(CORRF, p1, p2, outf);                                   // g_BA
  flow_est(CABV, p2, p1, outf + (size_t)Bn * 2 * 240 * 240);       // g_AB
}
